// Block_8297876816648
// MI455X (gfx1250) — matrix operand order repaired and checked by reading, not compiled, not run on hardware
//
#include <hip/hip_runtime.h>
#include <stdint.h>

// ---------------------------------------------------------------------------
// GPT-2 transformer block forward for MI455X (gfx1250, wave32, WMMA).
// bf16 WMMA for all matmuls; double-buffered LDS pipelines; async-to-LDS
// (GLOBAL_LOAD_ASYNC_TO_LDS_B128 + s_wait_asynccnt) for GEMM A-tile staging;
// flash attention with WMMA QK^T / PV; fused bias/GELU/LayerNorm epilogues.
// ---------------------------------------------------------------------------

#define S_CTX  2048
#define DMODEL 1024
#define NHEAD  16
#define BATCH  4
#define DFF    4096
#define MROWS  (BATCH * S_CTX)     // 8192
#define QKV_LD (3 * DMODEL)        // 3072

typedef unsigned short u16;
typedef __attribute__((ext_vector_type(16))) __bf16 v16bf;
typedef __attribute__((ext_vector_type(8)))  float  v8f;

union Frag16 { uint4 q[2]; v16bf v; u16 u[16]; };
union Half16 { uint4 q; u16 u[8]; };

__device__ __forceinline__ u16 f32_to_bf16(float f) {
  union { float f; unsigned u; } c; c.f = f;
  unsigned r = c.u + 0x7FFFu + ((c.u >> 16) & 1u);   // round-to-nearest-even
  return (u16)(r >> 16);
}

// Async copy of 32 bytes/lane from global to LDS (two B128 transfers).
// INST_OFFSET applies to both the global and LDS address (ISA 08 §4.4).
__device__ __forceinline__ void async_copy32(unsigned lds_off, const u16* g) {
  asm volatile("global_load_async_to_lds_b128 %0, %1, off"
               :: "v"(lds_off), "v"(g) : "memory");
  asm volatile("global_load_async_to_lds_b128 %0, %1, off offset:16"
               :: "v"(lds_off), "v"(g) : "memory");
}
__device__ __forceinline__ void wait_async() {
  asm volatile("s_wait_asynccnt 0x0" ::: "memory");
}

// --------------------------- fp32 -> bf16 convert ---------------------------
__global__ void cvt_f32_bf16_kernel(const float* __restrict__ in,
                                    u16* __restrict__ out, int n) {
  int i = blockIdx.x * blockDim.x + threadIdx.x;
  if (i < n) out[i] = f32_to_bf16(in[i]);
}

// ------------------------------- tiled GEMM --------------------------------
// C[M,N] = A[M,K](bf16) * W[K,N](bf16) + bias, optional GELU,
// write f32 and/or bf16. Block: 256 thr (8 waves), tile 128x128, kstep 32,
// double-buffered LDS; A tile staged via async-to-LDS DMA, W tile transposed
// through VGPRs with packed b32 LDS stores.
template<bool GELU, bool WRITE_B16, bool WRITE_F32>
__global__ void __launch_bounds__(256)
gemm_bf16(const u16* __restrict__ A, const u16* __restrict__ W,
          const float* __restrict__ bias,
          u16* __restrict__ outB, float* __restrict__ outF,
          int M, int N, int K)
{
  __shared__ u16 sA[2][128 * 40];   // 128 rows x 32 k, padded stride 40
  __shared__ u16 sB[2][128 * 40];   // Bt: 128 n-rows x 32 k, padded stride 40

  const int tid  = threadIdx.x;
  const int lane = tid & 31, wave = tid >> 5;
  const int hi = lane >> 4, li = lane & 15;
  const int wm = (wave & 1) * 64;        // wave M offset in tile
  const int wn = (wave >> 1) * 32;       // wave N offset in tile
  const int blockM = blockIdx.y * 128;
  const int blockN = blockIdx.x * 128;

  // staging maps: A -> (row, 16-elem segment); W -> (even k pair, 8 n cols)
  const int arow = tid >> 1,        aseg  = (tid & 1) * 16;
  const int bk2  = (tid >> 4) << 1, bnseg = (tid & 15) * 8;

  const u16* gA = A + (size_t)(blockM + arow) * K + aseg;
  const u16* gW = W + (size_t)bk2 * N + blockN + bnseg;
  const unsigned ldsA[2] = {
    (unsigned)(size_t)(void*)&sA[0][arow * 40 + aseg],
    (unsigned)(size_t)(void*)&sA[1][arow * 40 + aseg]
  };
  const int ntiles = K >> 5;

  v8f acc[4][2];
#pragma unroll
  for (int mt = 0; mt < 4; mt++)
#pragma unroll
    for (int nt = 0; nt < 2; nt++)
#pragma unroll
      for (int v = 0; v < 8; v++) acc[mt][nt][v] = 0.0f;

  // ---- prologue: stage tile 0 into buffers [0] ----
  {
    async_copy32(ldsA[0], gA);
    Half16 w0, w1;
    w0.q = *reinterpret_cast<const uint4*>(gW);
    w1.q = *reinterpret_cast<const uint4*>(gW + N);
#pragma unroll
    for (int i = 0; i < 8; i++) {
      unsigned pk = (unsigned)w0.u[i] | ((unsigned)w1.u[i] << 16);
      *reinterpret_cast<unsigned*>(&sB[0][(bnseg + i) * 40 + bk2]) = pk;
    }
  }
  wait_async();
  __syncthreads();

  for (int kt = 0; kt < ntiles; kt++) {
    const int cur = kt & 1, nxt = cur ^ 1;
    const bool has_next = (kt + 1) < ntiles;

    Half16 w0, w1;
    if (has_next) {                      // prefetch next tile under WMMAs
      const u16* ga = gA + (kt + 1) * 32;
      async_copy32(ldsA[nxt], ga);
      const u16* gw = gW + (size_t)(kt + 1) * 32 * N;
      w0.q = *reinterpret_cast<const uint4*>(gw);
      w1.q = *reinterpret_cast<const uint4*>(gw + N);
    }

    Frag16 af[4], bf2[2];
#pragma unroll
    for (int mt = 0; mt < 4; mt++) {
      const u16* p = &sA[cur][(wm + mt * 16 + li) * 40 + hi * 8];
      af[mt].q[0] = *reinterpret_cast<const uint4*>(p);
      af[mt].q[1] = *reinterpret_cast<const uint4*>(p + 16);
    }
#pragma unroll
    for (int nt = 0; nt < 2; nt++) {
      const u16* p = &sB[cur][(wn + nt * 16 + li) * 40 + hi * 8];
      bf2[nt].q[0] = *reinterpret_cast<const uint4*>(p);
      bf2[nt].q[1] = *reinterpret_cast<const uint4*>(p + 16);
    }
#pragma unroll
    for (int mt = 0; mt < 4; mt++)
#pragma unroll
      for (int nt = 0; nt < 2; nt++)
        acc[mt][nt] = __builtin_amdgcn_wmma_f32_16x16x32_bf16(
            false, af[mt].v, false, bf2[nt].v, (short)0, acc[mt][nt],
            false, false);

    if (has_next) {                      // transposed W store, packed b32
#pragma unroll
      for (int i = 0; i < 8; i++) {
        unsigned pk = (unsigned)w0.u[i] | ((unsigned)w1.u[i] << 16);
        *reinterpret_cast<unsigned*>(&sB[nxt][(bnseg + i) * 40 + bk2]) = pk;
      }
    }
    wait_async();
    __syncthreads();
  }

  // Epilogue: bias (+GELU), scattered stores
#pragma unroll
  for (int mt = 0; mt < 4; mt++) {
    int row = blockM + wm + mt * 16 + 8 * hi;
#pragma unroll
    for (int nt = 0; nt < 2; nt++) {
      int col = blockN + wn + nt * 16 + li;
      float bc = bias[col];
#pragma unroll
      for (int v = 0; v < 8; v++) {
        float val = acc[mt][nt][v] + bc;
        if (GELU) {
          float x3 = val * val * val;
          val = 0.5f * val *
                (1.0f + tanhf(0.7978845608028654f * (val + 0.044715f * x3)));
        }
        size_t idx = (size_t)(row + v) * N + col;
        if (WRITE_F32) outF[idx] = val;
        if (WRITE_B16) outB[idx] = f32_to_bf16(val);
      }
    }
  }
}

// --------------------------- flash attention -------------------------------
// Block: 128 thr = 4 waves; each wave owns a 16-row q tile (block covers 64
// q rows of one (b,h)). K blocks of 32 columns; online softmax; causal.
// V^T tile double-buffered in LDS (one barrier/iter); K frags straight from
// global (its [s][hd] layout IS the WMMA B layout) with all 8 B128 loads
// issued ahead of the WMMA chain; next K tile prefetched (global_prefetch_b8).
__global__ void __launch_bounds__(128)
attn_kernel(const u16* __restrict__ qkv, u16* __restrict__ out)
{
  __shared__ u16 sVt[2][64 * 40];   // V^T tile: [hd=64][k=32], stride 40
  __shared__ u16 sP[4][16 * 32];    // per-wave P scratch (C->A transpose)

  const int tid  = threadIdx.x;
  const int lane = tid & 31, wave = tid >> 5;
  const int hi = lane >> 4, li = lane & 15;
  const int bh = blockIdx.y, b = bh >> 4, h = bh & 15;
  const int q0 = blockIdx.x * 64;

  const size_t baseQ = ((size_t)b * S_CTX) * QKV_LD + (size_t)h * 64;
  const u16* qp = qkv + baseQ;
  const u16* kp = qkv + baseQ + DMODEL;
  const u16* vp = qkv + baseQ + 2 * DMODEL;

  // V staging map: even k pair, 8 hd cols per thread
  const int vk2 = (tid >> 3) << 1, vseg = (tid & 7) * 8;
  const u16* gV = vp + (size_t)vk2 * QKV_LD + vseg;

  // Q fragments (A operand, 16x64 -> two k-steps of 32), loaded once
  Frag16 qf[2];
  {
    const u16* p = qp + (size_t)(q0 + wave * 16 + li) * QKV_LD;
#pragma unroll
    for (int kk = 0; kk < 2; kk++) {
      qf[kk].q[0] = *reinterpret_cast<const uint4*>(p + kk * 32 + hi * 8);
      qf[kk].q[1] = *reinterpret_cast<const uint4*>(p + kk * 32 + hi * 8 + 16);
    }
  }

  v8f o[4];
  float m[8], l[8];
#pragma unroll
  for (int j = 0; j < 4; j++)
#pragma unroll
    for (int v = 0; v < 8; v++) o[j][v] = 0.0f;
#pragma unroll
  for (int v = 0; v < 8; v++) { m[v] = -1e30f; l[v] = 0.0f; }

  // ---- prologue: stage V tile 0 (transposed, packed b32 stores) ----
  {
    Half16 w0, w1;
    w0.q = *reinterpret_cast<const uint4*>(gV);
    w1.q = *reinterpret_cast<const uint4*>(gV + QKV_LD);
#pragma unroll
    for (int i = 0; i < 8; i++) {
      unsigned pk = (unsigned)w0.u[i] | ((unsigned)w1.u[i] << 16);
      *reinterpret_cast<unsigned*>(&sVt[0][(vseg + i) * 40 + vk2]) = pk;
    }
  }
  __syncthreads();

  const int kb_end = q0 + 64;            // causal: cover cols <= q0+63
  for (int kb = 0; kb < kb_end; kb += 32) {
    const int cur = (kb >> 5) & 1;
    const bool has_next = (kb + 32) < kb_end;

    Half16 w0, w1;
    if (has_next) {                       // next V tile + K prefetch
      const u16* g = gV + (size_t)(kb + 32) * QKV_LD;
      w0.q = *reinterpret_cast<const uint4*>(g);
      w1.q = *reinterpret_cast<const uint4*>(g + QKV_LD);
      __builtin_prefetch(kp + (size_t)(kb + 32 + li) * QKV_LD, 0, 3);
    }

    // Scores: issue ALL K-fragment loads (8x B128) before the WMMA chain so
    // the scheduler can overlap loads with matrix ops (partial loadcnt waits).
    Frag16 kf[2][2];                     // [nt][kk]
#pragma unroll
    for (int kk = 0; kk < 2; kk++) {
      const u16* p0 = kp + (size_t)(kb + li) * QKV_LD + kk * 32 + hi * 8;
      const u16* p1 = kp + (size_t)(kb + 16 + li) * QKV_LD + kk * 32 + hi * 8;
      kf[0][kk].q[0] = *reinterpret_cast<const uint4*>(p0);
      kf[0][kk].q[1] = *reinterpret_cast<const uint4*>(p0 + 16);
      kf[1][kk].q[0] = *reinterpret_cast<const uint4*>(p1);
      kf[1][kk].q[1] = *reinterpret_cast<const uint4*>(p1 + 16);
    }
    v8f s0, s1;
#pragma unroll
    for (int v = 0; v < 8; v++) { s0[v] = 0.0f; s1[v] = 0.0f; }
    s0 = __builtin_amdgcn_wmma_f32_16x16x32_bf16(
        false, qf[0].v, false, kf[0][0].v, (short)0, s0, false, false);
    s1 = __builtin_amdgcn_wmma_f32_16x16x32_bf16(
        false, qf[0].v, false, kf[1][0].v, (short)0, s1, false, false);
    s0 = __builtin_amdgcn_wmma_f32_16x16x32_bf16(
        false, qf[1].v, false, kf[0][1].v, (short)0, s0, false, false);
    s1 = __builtin_amdgcn_wmma_f32_16x16x32_bf16(
        false, qf[1].v, false, kf[1][1].v, (short)0, s1, false, false);

    // Online softmax update (rows = v + 8*hi, cols across 16-lane half)
    float p0v[8], p1v[8];
    const int rowa = q0 + wave * 16 + 8 * hi;
#pragma unroll
    for (int v = 0; v < 8; v++) {
      float a0 = s0[v] * 0.125f, a1 = s1[v] * 0.125f;   // 1/sqrt(64)
      int r = rowa + v;
      if (kb + li > r)      a0 = -1e30f;
      if (kb + 16 + li > r) a1 = -1e30f;
      float mx = fmaxf(a0, a1);
      for (int d = 1; d < 16; d <<= 1) mx = fmaxf(mx, __shfl_xor(mx, d));
      float nm = fmaxf(m[v], mx);
      float e0 = __expf(a0 - nm), e1 = __expf(a1 - nm);
      float rs = e0 + e1;
      for (int d = 1; d < 16; d <<= 1) rs += __shfl_xor(rs, d);
      float corr = __expf(m[v] - nm);
      l[v] = l[v] * corr + rs;
      m[v] = nm;
#pragma unroll
      for (int j = 0; j < 4; j++) o[j][v] *= corr;
      p0v[v] = e0; p1v[v] = e1;
    }

    // C-layout -> A-layout transpose of P through per-wave LDS
    {
      u16* w = sP[wave];
#pragma unroll
      for (int v = 0; v < 8; v++) {
        int r = v + 8 * hi;
        w[r * 32 + li]      = f32_to_bf16(p0v[v]);
        w[r * 32 + 16 + li] = f32_to_bf16(p1v[v]);
      }
    }
    asm volatile("s_wait_dscnt 0x0" ::: "memory");  // wave-local LDS fence
    Frag16 pa;
    {
      const u16* p = sP[wave] + li * 32 + hi * 8;
      pa.q[0] = *reinterpret_cast<const uint4*>(p);
      pa.q[1] = *reinterpret_cast<const uint4*>(p + 16);
    }

    // PV: o[j] += P(16x32) * V(32x16) from current V^T buffer
#pragma unroll
    for (int j = 0; j < 4; j++) {
      Frag16 vb;
      const u16* p = &sVt[cur][(j * 16 + li) * 40 + hi * 8];
      vb.q[0] = *reinterpret_cast<const uint4*>(p);
      vb.q[1] = *reinterpret_cast<const uint4*>(p + 16);
      o[j] = __builtin_amdgcn_wmma_f32_16x16x32_bf16(
          false, pa.v, false, vb.v, (short)0, o[j], false, false);
    }

    if (has_next) {                      // stage next V tile into other buffer
#pragma unroll
      for (int i = 0; i < 8; i++) {
        unsigned pk = (unsigned)w0.u[i] | ((unsigned)w1.u[i] << 16);
        *reinterpret_cast<unsigned*>(&sVt[cur ^ 1][(vseg + i) * 40 + vk2]) = pk;
      }
    }
    __syncthreads();
  }

  // Normalize and write merged-head bf16 output
  const size_t obase = ((size_t)b * S_CTX) * DMODEL + (size_t)h * 64;
#pragma unroll
  for (int v = 0; v < 8; v++) {
    float inv = 1.0f / l[v];
    int row = q0 + wave * 16 + v + 8 * hi;
#pragma unroll
    for (int j = 0; j < 4; j++)
      out[obase + (size_t)row * DMODEL + j * 16 + li] =
          f32_to_bf16(o[j][v] * inv);
  }
}

// --------------------------- fused add + LayerNorm -------------------------
template<bool WRITE_B16>
__global__ void __launch_bounds__(256)
add_ln_kernel(const float* __restrict__ X, const float* __restrict__ A,
              const float* __restrict__ g, const float* __restrict__ bb,
              float* __restrict__ outF, u16* __restrict__ outB)
{
  __shared__ float red[256];
  const int row = blockIdx.x, tid = threadIdx.x;
  const size_t base = (size_t)row * DMODEL;

  float vals[4], s = 0.0f;
#pragma unroll
  for (int i = 0; i < 4; i++) {
    int c = tid + i * 256;
    float v = X[base + c] + A[base + c];
    vals[i] = v; s += v;
  }
  red[tid] = s; __syncthreads();
  for (int off = 128; off > 0; off >>= 1) {
    if (tid < off) red[tid] += red[tid + off];
    __syncthreads();
  }
  float mean = red[0] * (1.0f / DMODEL);
  __syncthreads();

  float s2 = 0.0f;
#pragma unroll
  for (int i = 0; i < 4; i++) { float d = vals[i] - mean; s2 += d * d; }
  red[tid] = s2; __syncthreads();
  for (int off = 128; off > 0; off >>= 1) {
    if (tid < off) red[tid] += red[tid + off];
    __syncthreads();
  }
  float rstd = rsqrtf(red[0] * (1.0f / DMODEL) + 1e-5f);

#pragma unroll
  for (int i = 0; i < 4; i++) {
    int c = tid + i * 256;
    float y = g[c] * (vals[i] - mean) * rstd + bb[c];
    outF[base + c] = y;
    if (WRITE_B16) outB[base + c] = f32_to_bf16(y);
  }
}

// ------------------------------- launcher ----------------------------------
extern "C" void kernel_launch(void* const* d_in, const int* in_sizes, int n_in,
                              void* d_out, int out_size, void* d_ws,
                              size_t ws_size, hipStream_t stream)
{
  const float* x       = (const float*)d_in[0];
  const float* w_attn  = (const float*)d_in[1];
  const float* b_attn  = (const float*)d_in[2];
  const float* w_aproj = (const float*)d_in[3];
  const float* b_aproj = (const float*)d_in[4];
  const float* g1      = (const float*)d_in[5];
  const float* b1      = (const float*)d_in[6];
  const float* w_fc    = (const float*)d_in[7];
  const float* b_fc    = (const float*)d_in[8];
  const float* w_mproj = (const float*)d_in[9];
  const float* b_mproj = (const float*)d_in[10];
  const float* g2      = (const float*)d_in[11];
  const float* b2      = (const float*)d_in[12];
  float* out = (float*)d_out;

  // Workspace layout with lifetime-based aliasing (~192 MB, L2-resident)
  char* ws = (char*)d_ws;
  size_t off = 0;
  auto alloc = [&](size_t bytes) -> char* {
    char* p = ws + off; off += (bytes + 255) & ~(size_t)255; return p;
  };
  u16*   buf0   = (u16*)alloc((size_t)MROWS * DMODEL * 2);   // xb, later n_b
  u16*   wqkvb  = (u16*)alloc((size_t)DMODEL * QKV_LD * 2);
  u16*   waprjb = (u16*)alloc((size_t)DMODEL * DMODEL * 2);
  u16*   wfcb   = (u16*)alloc((size_t)DMODEL * DFF * 2);
  u16*   wmprjb = (u16*)alloc((size_t)DFF * DMODEL * 2);
  u16*   bufQH  = (u16*)alloc((size_t)MROWS * DFF * 2);      // qkv_b, later h_b
  u16*   attnb  = (u16*)alloc((size_t)MROWS * DMODEL * 2);
  float* bufA   = (float*)alloc((size_t)MROWS * DMODEL * 4); // a_f32, later m
  float* n_f32  = (float*)alloc((size_t)MROWS * DMODEL * 4);
  (void)ws_size; (void)n_in; (void)in_sizes; (void)out_size;

  u16* xb = buf0;        u16* n_b = buf0;
  u16* qkvb = bufQH;     u16* h_b = bufQH;
  float* a_f32 = bufA;   float* m_f32 = bufA;

  // 1) fp32 -> bf16 conversions
  int n;
  n = MROWS * DMODEL;
  cvt_f32_bf16_kernel<<<(n + 255) / 256, 256, 0, stream>>>(x, xb, n);
  n = DMODEL * QKV_LD;
  cvt_f32_bf16_kernel<<<(n + 255) / 256, 256, 0, stream>>>(w_attn, wqkvb, n);
  n = DMODEL * DMODEL;
  cvt_f32_bf16_kernel<<<(n + 255) / 256, 256, 0, stream>>>(w_aproj, waprjb, n);
  n = DMODEL * DFF;
  cvt_f32_bf16_kernel<<<(n + 255) / 256, 256, 0, stream>>>(w_fc, wfcb, n);
  n = DFF * DMODEL;
  cvt_f32_bf16_kernel<<<(n + 255) / 256, 256, 0, stream>>>(w_mproj, wmprjb, n);

  // 2) qkv = x @ w_attn + b_attn  -> bf16
  gemm_bf16<false, true, false>
      <<<dim3(QKV_LD / 128, MROWS / 128), 256, 0, stream>>>(
          xb, wqkvb, b_attn, qkvb, nullptr, MROWS, QKV_LD, DMODEL);

  // 3) causal flash attention -> merged heads bf16
  attn_kernel<<<dim3(S_CTX / 64, BATCH * NHEAD), 128, 0, stream>>>(qkvb, attnb);

  // 4) a = attn @ w_aproj + b_aproj -> f32
  gemm_bf16<false, false, true>
      <<<dim3(DMODEL / 128, MROWS / 128), 256, 0, stream>>>(
          attnb, waprjb, b_aproj, nullptr, a_f32, MROWS, DMODEL, DMODEL);

  // 5) n = LN(x + a)  -> f32 (residual) + bf16 (next GEMM)
  add_ln_kernel<true><<<MROWS, 256, 0, stream>>>(x, a_f32, g1, b1, n_f32, n_b);

  // 6) h = gelu(n @ w_fc + b_fc) -> bf16
  gemm_bf16<true, true, false>
      <<<dim3(DFF / 128, MROWS / 128), 256, 0, stream>>>(
          n_b, wfcb, b_fc, h_b, nullptr, MROWS, DFF, DMODEL);

  // 7) m = h @ w_mproj + b_mproj -> f32
  gemm_bf16<false, false, true>
      <<<dim3(DMODEL / 128, MROWS / 128), 256, 0, stream>>>(
          h_b, wmprjb, b_mproj, nullptr, m_f32, MROWS, DMODEL, DFF);

  // 8) out = LN(n + m)
  add_ln_kernel<false><<<MROWS, 256, 0, stream>>>(n_f32, m_f32, g2, b2,
                                                  out, nullptr);
}
